// Net_38431367365196
// MI455X (gfx1250) — matrix operand order repaired and checked by reading, not compiled, not run on hardware
//
#include <hip/hip_runtime.h>
#include <hip/hip_bf16.h>

typedef __attribute__((ext_vector_type(16))) _Float16 v16h;
typedef __attribute__((ext_vector_type(8)))  _Float16 v8h;
typedef __attribute__((ext_vector_type(8)))  float    v8f;

// ---------------- WMMA fragment loaders (contiguous-K order, same for A and B) ----------------
// A-matrix 16x32 f16: lanes 0-15 row=lane, K = k0..k0+15;
//                     lanes 16-31 row=lane-16, K = k0+16..k0+31.
__device__ inline v16h load_afrag(const _Float16* __restrict__ A, int lda, int row0, int k0) {
    int lane = threadIdx.x & 31;
    int r  = row0 + (lane & 15);
    int kk = k0 + ((lane >> 4) << 4);
    const _Float16* p = A + (long)r * lda + kk;
    v8h lo = *(const v8h*)p;          // K = kk .. kk+7
    v8h hi = *(const v8h*)(p + 8);    // K = kk+8 .. kk+15
    v16h a;
#pragma unroll
    for (int i = 0; i < 8; ++i) { a[i] = lo[i]; a[i + 8] = hi[i]; }
    return a;
}

// B-matrix 32x16 f16: lane col = lane&15; lanes 0-15 hold K=k0..k0+15,
// lanes 16-31 hold K=k0+16..k0+31, packed as 16 consecutive halves.
// colptr = start of this lane's contiguous K-run for its column.
__device__ inline v16h load_bfrag(const _Float16* __restrict__ colptr, int k0) {
    int lane = threadIdx.x & 31;
    return *(const v16h*)(colptr + k0 + ((lane >> 4) << 4));
}

// ---------------- gfx1250 async global->LDS copy (ASYNCcnt-tracked) ----------------
__device__ inline void async_ld16(unsigned lds_off, const void* g) {
    asm volatile("global_load_async_to_lds_b128 %0, %1, off"
                 :: "v"(lds_off), "v"((unsigned long long)g)
                 : "memory");
}
__device__ inline void wait_async0() {
    asm volatile("s_wait_asynccnt 0x0" ::: "memory");
}

// ---------------- Stage 1: global mean/var reduction (deterministic) ----------------
__global__ void k_reduce(const float* __restrict__ x, float* __restrict__ part) {
    __shared__ float s1[256], s2[256];
    int tid = threadIdx.x;
    long i0 = ((long)blockIdx.x * 256 + tid) * 4;
    float s = 0.f, ss = 0.f;
#pragma unroll
    for (int j = 0; j < 4; ++j) { float v = x[i0 + j]; s += v; ss += v * v; }
    s1[tid] = s; s2[tid] = ss; __syncthreads();
    for (int off = 128; off > 0; off >>= 1) {
        if (tid < off) { s1[tid] += s1[tid + off]; s2[tid] += s2[tid + off]; }
        __syncthreads();
    }
    if (tid == 0) { part[blockIdx.x] = s1[0]; part[512 + blockIdx.x] = s2[0]; }
}

__global__ void k_finalize(const float* __restrict__ part, float* __restrict__ stats) {
    __shared__ float s1[256], s2[256];
    int tid = threadIdx.x;
    s1[tid] = part[tid] + part[tid + 256];
    s2[tid] = part[512 + tid] + part[512 + tid + 256];
    __syncthreads();
    for (int off = 128; off > 0; off >>= 1) {
        if (tid < off) { s1[tid] += s1[tid + off]; s2[tid] += s2[tid + off]; }
        __syncthreads();
    }
    if (tid == 0) {
        float inv = 1.0f / 524288.0f;
        float mu  = s1[0] * inv;
        float var = s2[0] * inv - mu * mu;
        stats[2] = mu;
        stats[3] = rsqrtf(var + 1e-5f);
    }
}

// ---------------- Pack normalized tokens [B*N x C] f16 ----------------
__global__ void k_pack_xn(const float* __restrict__ x, const float* __restrict__ stats,
                          _Float16* __restrict__ xn) {
    int idx = blockIdx.x * 256 + threadIdx.x;   // 0..524287, layout [b][c][n]
    float mu = stats[2], rs = stats[3];
    int b = idx >> 16, rem = idx & 65535;
    int c = rem >> 10, n = rem & 1023;
    xn[(((long)b << 10) + n) * 64 + c] = (_Float16)((x[idx] - mu) * rs);
}

__global__ void k_f32_to_f16(const float* __restrict__ src, _Float16* __restrict__ dst, int n) {
    int i = blockIdx.x * 256 + threadIdx.x;
    if (i < n) dst[i] = (_Float16)src[i];
}

// ---------------- Direct 3x3 SAME conv for K/V feature maps ----------------
__global__ __launch_bounds__(256) void k_conv(const float* __restrict__ x,
        const float* __restrict__ wk, const float* __restrict__ bk,
        const float* __restrict__ wv, const float* __restrict__ bv,
        _Float16* __restrict__ kf, _Float16* __restrict__ vf) {
    __shared__ float plane[1024];
    int bc = blockIdx.x, b = bc >> 6, co = bc & 63, tid = threadIdx.x;
    const float* xb = x + (long)b * 64 * 1024;
    float acck[4] = {0,0,0,0}, accv[4] = {0,0,0,0};
    for (int ci = 0; ci < 64; ++ci) {
        __syncthreads();
#pragma unroll
        for (int j = 0; j < 4; ++j) plane[tid + 256 * j] = xb[ci * 1024 + tid + 256 * j];
        __syncthreads();
        float wkk[9], wvv[9];
#pragma unroll
        for (int t = 0; t < 9; ++t) {
            wkk[t] = wk[(co * 64 + ci) * 9 + t];
            wvv[t] = wv[(co * 64 + ci) * 9 + t];
        }
#pragma unroll
        for (int p = 0; p < 4; ++p) {
            int pix = tid * 4 + p, h = pix >> 5, w = pix & 31;
#pragma unroll
            for (int dy = 0; dy < 3; ++dy) {
                int hh = h + dy - 1;
                if (hh < 0 || hh > 31) continue;
#pragma unroll
                for (int dx = 0; dx < 3; ++dx) {
                    int ww = w + dx - 1;
                    if (ww < 0 || ww > 31) continue;
                    float v = plane[hh * 32 + ww];
                    acck[p] = fmaf(v, wkk[dy * 3 + dx], acck[p]);
                    accv[p] = fmaf(v, wvv[dy * 3 + dx], accv[p]);
                }
            }
        }
    }
    float bkc = bk[co], bvc = bv[co];
#pragma unroll
    for (int p = 0; p < 4; ++p) {
        int pix = tid * 4 + p;
        long o = ((long)b * 1024 + pix) * 64 + co;
        kf[o] = (_Float16)(acck[p] + bkc);
        vf[o] = (_Float16)(accv[p] + bvc);
    }
}

// ---------------- Q/K projection GEMM: out = (A[Mx64] @ W[512x64]^T + bias) * scale ----------------
__global__ __launch_bounds__(256) void k_proj_qk(const _Float16* __restrict__ A,
        const _Float16* __restrict__ W, const float* __restrict__ bias, float scale,
        _Float16* __restrict__ out) {
    int wave = threadIdx.x >> 5, lane = threadIdx.x & 31;
    int tile = blockIdx.x * 8 + wave;       // 512 m-tiles x 32 n-tiles
    int tm = tile >> 5, tn = tile & 31;
    v16h a0 = load_afrag(A, 64, tm * 16, 0);
    v16h a1 = load_afrag(A, 64, tm * 16, 32);
    const _Float16* colptr = W + (long)(tn * 16 + (lane & 15)) * 64;
    v8f c = {};
    c = __builtin_amdgcn_wmma_f32_16x16x32_f16(false, a0, false, load_bfrag(colptr, 0),  (short)0, c, false, false);
    c = __builtin_amdgcn_wmma_f32_16x16x32_f16(false, a1, false, load_bfrag(colptr, 32), (short)0, c, false, false);
    int col = tn * 16 + (lane & 15);
    float bb = bias[col];
#pragma unroll
    for (int i = 0; i < 8; ++i) {
        int row = tm * 16 + i + ((lane >> 4) << 3);
        out[(long)row * 512 + col] = (_Float16)((c[i] + bb) * scale);
    }
}

// ---------------- V projection GEMM with transposed store vT[B][H][64][N] ----------------
__global__ __launch_bounds__(256) void k_proj_v(const _Float16* __restrict__ A,
        const _Float16* __restrict__ W, const float* __restrict__ bias,
        _Float16* __restrict__ out) {
    int wave = threadIdx.x >> 5, lane = threadIdx.x & 31;
    int tile = blockIdx.x * 8 + wave;
    int tm = tile >> 5, tn = tile & 31;
    v16h a0 = load_afrag(A, 64, tm * 16, 0);
    v16h a1 = load_afrag(A, 64, tm * 16, 32);
    const _Float16* colptr = W + (long)(tn * 16 + (lane & 15)) * 64;
    v8f c = {};
    c = __builtin_amdgcn_wmma_f32_16x16x32_f16(false, a0, false, load_bfrag(colptr, 0),  (short)0, c, false, false);
    c = __builtin_amdgcn_wmma_f32_16x16x32_f16(false, a1, false, load_bfrag(colptr, 32), (short)0, c, false, false);
    int col = tn * 16 + (lane & 15);
    int h = col >> 6, d = col & 63;
    float bb = bias[col];
#pragma unroll
    for (int i = 0; i < 8; ++i) {
        int row = tm * 16 + i + ((lane >> 4) << 3);
        int bidx = row >> 10, n = row & 1023;
        out[(((long)(bidx * 8 + h) * 64 + d) << 10) + n] = (_Float16)(c[i] + bb);
    }
}

// ---------------- Flash attention with async global->LDS staging ----------------
// 1 block = one (b,h,128-query chunk); 8 waves x 16 query rows.
// K tile (32x64 f16), vT tile (64x32 f16) shared block-wide; bias tile per wave.
__global__ __launch_bounds__(256) void k_attn(const _Float16* __restrict__ q,
        const _Float16* __restrict__ k, const _Float16* __restrict__ vT,
        const float* __restrict__ biasB, _Float16* __restrict__ res) {
    __shared__ __align__(16) _Float16 lds_p[8][16 * 32];   // 8 KB  P staging
    __shared__ __align__(16) _Float16 ktile[32 * 64];      // 4 KB
    __shared__ __align__(16) _Float16 vtile[64 * 32];      // 4 KB
    __shared__ __align__(16) float    btile[8][16 * 32];   // 16 KB bias tiles
    int tid = threadIdx.x;
    int wave = tid >> 5, lane = tid & 31;
    int qc = blockIdx.x & 7, bh = blockIdx.x >> 3;
    int b = bh >> 3, h = bh & 7;
    int q0 = qc * 128 + wave * 16;
    int lanecol = lane & 15, rowoff = (lane >> 4) << 3;

    v16h a0 = load_afrag(q, 512, b * 1024 + q0, h * 64);
    v16h a1 = load_afrag(q, 512, b * 1024 + q0, h * 64 + 32);

    const float*    bptr = biasB + (long)h * 1024 * 1024;
    const _Float16* kb   = k  + (long)b * 1024 * 512 + h * 64;
    const _Float16* vb   = vT + (long)(b * 8 + h) * 64 * 1024;

    // async-copy address precompute
    unsigned        k_lds   = (unsigned)(size_t)&ktile[(tid >> 3) * 64 + (tid & 7) * 8];
    const _Float16* kg_base = kb + (long)(tid >> 3) * 512 + (tid & 7) * 8;
    unsigned        v_lds   = (unsigned)(size_t)&vtile[(tid >> 2) * 32 + (tid & 3) * 8];
    const _Float16* vg_base = vb + (long)(tid >> 2) * 1024 + (tid & 3) * 8;
    unsigned     b_lds[4];
    const float* bg_base[4];
#pragma unroll
    for (int i = 0; i < 4; ++i) {
        int f = lane * 16 + i * 4;                 // flat float idx in 16x32 tile
        b_lds[i]   = (unsigned)(size_t)&btile[wave][f];
        bg_base[i] = bptr + (long)(q0 + (f >> 5)) * 1024 + (f & 31);
    }

    v8f acc0 = {}, acc1 = {}, acc2 = {}, acc3 = {};
    float m[8], l[8];
#pragma unroll
    for (int i = 0; i < 8; ++i) { m[i] = -1e30f; l[i] = 0.f; }

    for (int j0 = 0; j0 < 1024; j0 += 32) {
        __syncthreads();                       // prev-iter tile reads complete
        async_ld16(k_lds, kg_base + (long)j0 * 512);
        async_ld16(v_lds, vg_base + j0);
#pragma unroll
        for (int i = 0; i < 4; ++i) async_ld16(b_lds[i], bg_base[i] + j0);
        wait_async0();
        __syncthreads();                       // tiles visible to all waves

        // scores: S[16x32] via 4 WMMA from LDS K tile
        const _Float16* kc0 = ktile + lanecol * 64;
        const _Float16* kc1 = ktile + (16 + lanecol) * 64;
        v8f s0 = {}, s1 = {};
        s0 = __builtin_amdgcn_wmma_f32_16x16x32_f16(false, a0, false, load_bfrag(kc0, 0),  (short)0, s0, false, false);
        s0 = __builtin_amdgcn_wmma_f32_16x16x32_f16(false, a1, false, load_bfrag(kc0, 32), (short)0, s0, false, false);
        s1 = __builtin_amdgcn_wmma_f32_16x16x32_f16(false, a0, false, load_bfrag(kc1, 0),  (short)0, s1, false, false);
        s1 = __builtin_amdgcn_wmma_f32_16x16x32_f16(false, a1, false, load_bfrag(kc1, 32), (short)0, s1, false, false);

        // online softmax
#pragma unroll
        for (int i = 0; i < 8; ++i) {
            int rr = i + rowoff;
            float v0 = s0[i] + btile[wave][rr * 32 + lanecol];
            float v1 = s1[i] + btile[wave][rr * 32 + 16 + lanecol];
            float mx = fmaxf(v0, v1);
            mx = fmaxf(mx, __shfl_xor(mx, 1));
            mx = fmaxf(mx, __shfl_xor(mx, 2));
            mx = fmaxf(mx, __shfl_xor(mx, 4));
            mx = fmaxf(mx, __shfl_xor(mx, 8));
            float mnew = fmaxf(m[i], mx);
            float sc = __expf(m[i] - mnew);
            m[i] = mnew;
            float p0 = __expf(v0 - mnew);
            float p1 = __expf(v1 - mnew);
            float rs = p0 + p1;
            rs += __shfl_xor(rs, 1);
            rs += __shfl_xor(rs, 2);
            rs += __shfl_xor(rs, 4);
            rs += __shfl_xor(rs, 8);
            l[i] = l[i] * sc + rs;
            acc0[i] *= sc; acc1[i] *= sc; acc2[i] *= sc; acc3[i] *= sc;
            lds_p[wave][rr * 32 + lanecol]      = (_Float16)p0;
            lds_p[wave][rr * 32 + 16 + lanecol] = (_Float16)p1;
        }
        __syncthreads();                       // P staged

        v16h pf = load_afrag(&lds_p[wave][0], 32, 0, 0);
        const _Float16* vc = vtile + lanecol * 32;
        acc0 = __builtin_amdgcn_wmma_f32_16x16x32_f16(false, pf, false, load_bfrag(vc, 0),            (short)0, acc0, false, false);
        acc1 = __builtin_amdgcn_wmma_f32_16x16x32_f16(false, pf, false, load_bfrag(vc + 16 * 32, 0),  (short)0, acc1, false, false);
        acc2 = __builtin_amdgcn_wmma_f32_16x16x32_f16(false, pf, false, load_bfrag(vc + 32 * 32, 0),  (short)0, acc2, false, false);
        acc3 = __builtin_amdgcn_wmma_f32_16x16x32_f16(false, pf, false, load_bfrag(vc + 48 * 32, 0),  (short)0, acc3, false, false);
    }

#pragma unroll
    for (int i = 0; i < 8; ++i) {
        float inv = 1.0f / l[i];
        int row = b * 1024 + q0 + i + rowoff;
        long base = (long)row * 512 + h * 64;
        res[base + lanecol]      = (_Float16)(acc0[i] * inv);
        res[base + 16 + lanecol] = (_Float16)(acc1[i] * inv);
        res[base + 32 + lanecol] = (_Float16)(acc2[i] * inv);
        res[base + 48 + lanecol] = (_Float16)(acc3[i] * inv);
    }
}

// ---------------- Output projection: out = res[8192x512] @ wo[64x512]^T + bo + x ----------------
__global__ __launch_bounds__(256) void k_outproj(const _Float16* __restrict__ A,
        const _Float16* __restrict__ W, const float* __restrict__ bias,
        const float* __restrict__ x, float* __restrict__ out) {
    int wave = threadIdx.x >> 5, lane = threadIdx.x & 31;
    int tile = blockIdx.x * 8 + wave;      // 512 m-tiles x 4 n-tiles
    int tm = tile >> 2, tn = tile & 3;
    const _Float16* colptr = W + (long)(tn * 16 + (lane & 15)) * 512;
    v8f c = {};
    for (int k0 = 0; k0 < 512; k0 += 32) {
        v16h a = load_afrag(A, 512, tm * 16, k0);
        c = __builtin_amdgcn_wmma_f32_16x16x32_f16(false, a, false, load_bfrag(colptr, k0), (short)0, c, false, false);
    }
    int col = tn * 16 + (lane & 15);
    float bb = bias[col];
#pragma unroll
    for (int i = 0; i < 8; ++i) {
        int row = tm * 16 + i + ((lane >> 4) << 3);
        long idx = (long)row * 64 + col;   // flat [b][n][c] == flat [b][c][h][w]
        out[idx] = c[i] + bb + x[idx];
    }
}

extern "C" void kernel_launch(void* const* d_in, const int* in_sizes, int n_in,
                              void* d_out, int out_size, void* d_ws, size_t ws_size,
                              hipStream_t stream) {
    const float* x     = (const float*)d_in[0];
    const float* wkc   = (const float*)d_in[1];
    const float* bkc   = (const float*)d_in[2];
    const float* wvc   = (const float*)d_in[3];
    const float* bvc   = (const float*)d_in[4];
    const float* wq    = (const float*)d_in[5];
    const float* bq    = (const float*)d_in[6];
    const float* wk    = (const float*)d_in[7];
    const float* bk    = (const float*)d_in[8];
    const float* wv    = (const float*)d_in[9];
    const float* bv    = (const float*)d_in[10];
    const float* wo    = (const float*)d_in[11];
    const float* bo    = (const float*)d_in[12];
    const float* biasB = (const float*)d_in[13];
    float* out = (float*)d_out;

    char* ws = (char*)d_ws;
    size_t off = 0;
    auto carve = [&](size_t bytes) -> char* {
        char* p = ws + off;
        off = (off + bytes + 255) & ~(size_t)255;
        return p;
    };
    float*    stats = (float*)carve(64);                 // [2]=mu, [3]=rstd
    float*    part  = (float*)carve(1024 * 4);           // per-block partials
    _Float16* xn    = (_Float16*)carve(8192ull * 64 * 2);
    _Float16* kf    = (_Float16*)carve(8192ull * 64 * 2);
    _Float16* vf    = (_Float16*)carve(8192ull * 64 * 2);
    _Float16* qb    = (_Float16*)carve(8192ull * 512 * 2);
    _Float16* kb    = (_Float16*)carve(8192ull * 512 * 2);
    _Float16* vT    = (_Float16*)carve(8192ull * 512 * 2);
    _Float16* res   = (_Float16*)carve(8192ull * 512 * 2);
    _Float16* wq16  = (_Float16*)carve(32768ull * 2);
    _Float16* wk16  = (_Float16*)carve(32768ull * 2);
    _Float16* wv16  = (_Float16*)carve(32768ull * 2);
    _Float16* wo16  = (_Float16*)carve(32768ull * 2);

    k_reduce  <<<512, 256, 0, stream>>>(x, part);
    k_finalize<<<1,   256, 0, stream>>>(part, stats);
    k_pack_xn <<<2048,256, 0, stream>>>(x, stats, xn);
    k_f32_to_f16<<<128, 256, 0, stream>>>(wq, wq16, 32768);
    k_f32_to_f16<<<128, 256, 0, stream>>>(wk, wk16, 32768);
    k_f32_to_f16<<<128, 256, 0, stream>>>(wv, wv16, 32768);
    k_f32_to_f16<<<128, 256, 0, stream>>>(wo, wo16, 32768);
    k_conv    <<<512, 256, 0, stream>>>(x, wkc, bkc, wvc, bvc, kf, vf);

    k_proj_qk<<<2048, 256, 0, stream>>>(xn, wq16, bq, 0.125f, qb);  // 1/sqrt(64) folded into Q
    k_proj_qk<<<2048, 256, 0, stream>>>(kf, wk16, bk, 1.0f,   kb);
    k_proj_v <<<2048, 256, 0, stream>>>(vf, wv16, bv, vT);          // transposed store

    k_attn   <<<512, 256, 0, stream>>>(qb, kb, vT, biasB, res);
    k_outproj<<<256, 256, 0, stream>>>(res, wo16, bo, x, out);
}
